// SparseDiffAttention_32573031972981
// MI455X (gfx1250) — matrix operand order repaired and checked by reading, not compiled, not run on hardware
//
#include <hip/hip_runtime.h>

// ---------------------------------------------------------------------------
// SparseDiffAttention recompute step for MI455X (gfx1250, wave32, WMMA).
//   B=1, H=16, N=3072, D=128, QG=192 (M=16 groups), topk=512.
// Compute-bound (~9e10 FLOPs vs ~125MB traffic) -> f16 WMMA with f32 acc,
// flash-attention style (scores never hit HBM). cs colsums accumulated from
// the same score tiles. Top-k via threshold bisection; ~1% random columns
// approximated with a deterministic integer hash (JAX threefry not
// reproducible in-kernel; no runtime check in this loop).
// Workspace layout (needs ~44.5 MB):
//   qf16 | kf16 | vf16 | cs[256][3072] f32 | counts[256] i32 | idx[256][3072] i32
// ---------------------------------------------------------------------------

#define HD    128      // head dim
#define NN    3072     // sequence length
#define NH    16       // heads
#define QGRP  192      // query group size
#define NGRP  16       // groups per head (NN/QGRP)
#define WPB   12       // waves per block (one 16-row q tile per wave)
#define TPB   (WPB*32) // 384 threads
#define KTPAD 40       // LDS row stride for transposed K tile (16B aligned)
#define VPAD  136      // LDS row stride for V tile (16B aligned)

typedef _Float16 h16;
typedef __attribute__((ext_vector_type(16))) _Float16 v16h;
typedef __attribute__((ext_vector_type(8)))  _Float16 v8h;
typedef __attribute__((ext_vector_type(4)))  _Float16 v4h;
typedef __attribute__((ext_vector_type(8)))  float    v8f;
typedef __attribute__((ext_vector_type(4)))  float    v4f;

__device__ __forceinline__ v8f wmma_f16(v16h a, v16h b, v8f c) {
  return __builtin_amdgcn_wmma_f32_16x16x32_f16(
      /*neg_a=*/false, a, /*neg_b=*/false, b,
      /*c_mod=*/(short)0, c, /*reuse_a=*/false, /*reuse_b=*/false);
}

__device__ __forceinline__ v16h cat16(v8h lo, v8h hi) {
  v16h r;
#pragma unroll
  for (int e = 0; e < 8; ++e) { r[e] = lo[e]; r[8 + e] = hi[e]; }
  return r;
}

// ---------------------------------------------------------------- convert ---
__global__ void k_convert(const float* __restrict__ q, const float* __restrict__ k,
                          const float* __restrict__ v,
                          h16* __restrict__ qb, h16* __restrict__ kb,
                          h16* __restrict__ vb, int n4) {
  int i = blockIdx.x * blockDim.x + threadIdx.x;
  if (i >= n4) return;
  const float SC = 0.08838834764831845f;  // 1/sqrt(128), folded into Q
  v4f qv = *(const v4f*)(q + (size_t)i * 4);
  v4f kv = *(const v4f*)(k + (size_t)i * 4);
  v4f vv = *(const v4f*)(v + (size_t)i * 4);
  v4h qo, ko, vo;
#pragma unroll
  for (int j = 0; j < 4; ++j) {
    qo[j] = (h16)(qv[j] * SC);
    ko[j] = (h16)kv[j];
    vo[j] = (h16)vv[j];
  }
  *(v4h*)(qb + (size_t)i * 4) = qo;
  *(v4h*)(kb + (size_t)i * 4) = ko;
  *(v4h*)(vb + (size_t)i * 4) = vo;
}

__global__ void k_zero_counts(int* __restrict__ counts) {
  counts[threadIdx.x] = 0;
}

// ------------------------------------------------------- dense pass + cs ----
__global__ __launch_bounds__(TPB) void k_pass1(
    const h16* __restrict__ qb, const h16* __restrict__ kb,
    const h16* __restrict__ vb, const float* __restrict__ prev_l,
    float* __restrict__ out, float* __restrict__ cs_g) {
  __shared__ h16   sKt[HD * KTPAD];     // K tile transposed [dim][key]
  __shared__ h16   sV[32 * VPAD];       // V tile [key][dim]
  __shared__ h16   sP[WPB * 512];       // per-wave 16x32 P tiles
  __shared__ float sCs[NN];

  const int row_id = blockIdx.x;             // h*NGRP + m
  const int h = row_id / NGRP, m = row_id % NGRP;
  const int qbase = m * QGRP;
  const int tid = threadIdx.x;
  const int w = tid >> 5, lane = tid & 31;
  const int half = lane >> 4, l16 = lane & 15;
  const int o1 = 8 * half, o2 = 16 + 8 * half;  // A-layout K offsets per half

  for (int i = tid; i < NN; i += TPB) sCs[i] = 0.f;

  // Q fragments (A layout: lanes 0-15 hold M=lane, K in {0..7,16..23}).
  const long qrow = (long)h * NN + qbase + w * 16 + l16;
  v16h aQ[4];
#pragma unroll
  for (int ks = 0; ks < 4; ++ks) {
    v8h lo = *(const v8h*)(qb + qrow * HD + ks * 32 + o1);
    v8h hi = *(const v8h*)(qb + qrow * HD + ks * 32 + o2);
    aQ[ks] = cat16(lo, hi);
  }
  // prev_l for the 8 C-matrix rows this lane owns (row = v + 8*half).
  float pl[8];
#pragma unroll
  for (int v = 0; v < 8; ++v)
    pl[v] = prev_l[(long)h * NN + qbase + w * 16 + v + 8 * half];

  v8f oacc[8];
#pragma unroll
  for (int dt = 0; dt < 8; ++dt) oacc[dt] = (v8f){};
  float mrun[8], lrun[8];
#pragma unroll
  for (int v = 0; v < 8; ++v) { mrun[v] = -3.0e38f; lrun[v] = 0.f; }

  __syncthreads();

  for (int kb0 = 0; kb0 < NN; kb0 += 32) {
    // Stage K^T and V tiles (f16, 16B chunks).
    for (int c = tid; c < 512; c += TPB) {
      int key = c >> 4, d0 = (c & 15) * 8;
      v8h kv = *(const v8h*)(kb + ((long)h * NN + kb0 + key) * HD + d0);
#pragma unroll
      for (int j = 0; j < 8; ++j) sKt[(d0 + j) * KTPAD + key] = kv[j];
      v8h vv = *(const v8h*)(vb + ((long)h * NN + kb0 + key) * HD + d0);
      *(v8h*)(sV + key * VPAD + d0) = vv;
    }
    __syncthreads();

    // S = Q @ K^T : two 16-col tiles, K-dim chained over 4 slices.
    v8f c0 = (v8f){}, c1 = (v8f){};
#pragma unroll
    for (int ks = 0; ks < 4; ++ks) {
      const h16* kp = sKt + ks * 32 * KTPAD + l16;  // B layout: n = l16, k in A order
      v16h bK0, bK1;
#pragma unroll
      for (int e = 0; e < 8; ++e) {
        bK0[e] = kp[(o1 + e) * KTPAD];      bK0[8 + e] = kp[(o2 + e) * KTPAD];
        bK1[e] = kp[(o1 + e) * KTPAD + 16]; bK1[8 + e] = kp[(o2 + e) * KTPAD + 16];
      }
      c0 = wmma_f16(aQ[ks], bK0, c0);
      c1 = wmma_f16(aQ[ks], bK1, c1);
    }

    // Online softmax; row data lives across the 16 lanes of each half.
    float al[8], p0[8], p1[8];
    float cs0 = 0.f, cs1 = 0.f;
#pragma unroll
    for (int v = 0; v < 8; ++v) {
      float t = fmaxf(c0[v], c1[v]);
      t = fmaxf(t, __shfl_xor(t, 1, 32));
      t = fmaxf(t, __shfl_xor(t, 2, 32));
      t = fmaxf(t, __shfl_xor(t, 4, 32));
      t = fmaxf(t, __shfl_xor(t, 8, 32));
      float mn = fmaxf(mrun[v], t);
      al[v] = __expf(mrun[v] - mn);
      mrun[v] = mn;
      p0[v] = __expf(c0[v] - mn);
      p1[v] = __expf(c1[v] - mn);
      float rs = p0[v] + p1[v];
      rs += __shfl_xor(rs, 1, 32);
      rs += __shfl_xor(rs, 2, 32);
      rs += __shfl_xor(rs, 4, 32);
      rs += __shfl_xor(rs, 8, 32);
      lrun[v] = lrun[v] * al[v] + rs;
      // colsum terms for cs: exp(s - prev_l[q])  (no max trick needed)
      cs0 += __expf(c0[v] - pl[v]);
      cs1 += __expf(c1[v] - pl[v]);
    }
#pragma unroll
    for (int dt = 0; dt < 8; ++dt)
#pragma unroll
      for (int v = 0; v < 8; ++v) oacc[dt][v] *= al[v];

    // cs: fold the two row-halves, one LDS atomic per lane.
    cs0 += __shfl_xor(cs0, 16, 32);
    cs1 += __shfl_xor(cs1, 16, 32);
    if (half == 0) atomicAdd(&sCs[kb0 + l16], cs0);
    else           atomicAdd(&sCs[kb0 + 16 + l16], cs1);

    // Re-layout P (C layout -> A layout) through per-wave LDS.
    h16* pw = sP + w * 512;
#pragma unroll
    for (int v = 0; v < 8; ++v) {
      pw[(v + 8 * half) * 32 + l16]      = (h16)p0[v];
      pw[(v + 8 * half) * 32 + 16 + l16] = (h16)p1[v];
    }
    const h16* pp = pw + l16 * 32;
    v16h aP = cat16(*(const v8h*)(pp + o1), *(const v8h*)(pp + o2));

    // O += P @ V  (8 d-tiles)
#pragma unroll
    for (int dt = 0; dt < 8; ++dt) {
      const h16* vp = sV + dt * 16 + l16;     // B layout: n = l16, k in A order
      v16h bV;
#pragma unroll
      for (int e = 0; e < 8; ++e) { bV[e] = vp[(o1 + e) * VPAD]; bV[8 + e] = vp[(o2 + e) * VPAD]; }
      oacc[dt] = wmma_f16(aP, bV, oacc[dt]);
    }
    __syncthreads();
  }

  // Epilogue: o = O / l
  float inv[8];
#pragma unroll
  for (int v = 0; v < 8; ++v) inv[v] = 1.f / lrun[v];
#pragma unroll
  for (int dt = 0; dt < 8; ++dt)
#pragma unroll
    for (int v = 0; v < 8; ++v) {
      long r = (long)h * NN + qbase + w * 16 + v + 8 * half;
      out[r * HD + dt * 16 + l16] = oacc[dt][v] * inv[v];
    }
  __syncthreads();
  for (int i = tid; i < NN; i += TPB)
    cs_g[(long)row_id * NN + i] = sCs[i];
}

// -------------------------------------------- top-k threshold + compact -----
__global__ __launch_bounds__(256) void k_select(
    const float* __restrict__ cs_g, const int* __restrict__ topk_p,
    int* __restrict__ counts, int* __restrict__ idx_g) {
  __shared__ int scnt;
  const int row = blockIdx.x;
  const int tid = threadIdx.x;
  const int K = topk_p[0];
  unsigned u[12];
#pragma unroll
  for (int j = 0; j < 12; ++j)
    u[j] = __float_as_uint(cs_g[(long)row * NN + tid * 12 + j]);

  // cs >= 0 so uint order == float order: bisect the threshold bits.
  unsigned thr = 0;
  for (int bit = 30; bit >= 0; --bit) {
    unsigned cand = thr | (1u << bit);
    if (tid == 0) scnt = 0;
    __syncthreads();
    int c = 0;
#pragma unroll
    for (int j = 0; j < 12; ++j) c += (u[j] >= cand) ? 1 : 0;
    atomicAdd(&scnt, c);
    __syncthreads();
    if (scnt >= K) thr = cand;
    __syncthreads();
  }

  // mask = top-k | ~1% hashed "random" columns; compact (order-free).
#pragma unroll
  for (int j = 0; j < 12; ++j) {
    int col = tid * 12 + j;
    unsigned x = (unsigned)(row * NN + col) * 2654435761u;
    x ^= x >> 16; x *= 0x7feb352du; x ^= x >> 15; x *= 0x846ca68bu; x ^= x >> 16;
    if (u[j] >= thr || (x % 100u) == 0u) {
      int pos = atomicAdd(&counts[row], 1);
      idx_g[(long)row * NN + pos] = col;
    }
  }
}

// ------------------------------------------------------- sparse pass --------
__global__ __launch_bounds__(TPB) void k_sparse(
    const h16* __restrict__ qb, const h16* __restrict__ kb,
    const h16* __restrict__ vb, const int* __restrict__ counts,
    const int* __restrict__ idx_g, const float* __restrict__ o_dense,
    float* __restrict__ out_cache) {
  __shared__ h16 sKt[HD * KTPAD];
  __shared__ h16 sV[32 * VPAD];
  __shared__ h16 sP[WPB * 512];
  __shared__ int sCol[32];

  const int row_id = blockIdx.x;
  const int h = row_id / NGRP, m = row_id % NGRP;
  const int qbase = m * QGRP;
  const int tid = threadIdx.x;
  const int w = tid >> 5, lane = tid & 31;
  const int half = lane >> 4, l16 = lane & 15;
  const int o1 = 8 * half, o2 = 16 + 8 * half;

  const int count = counts[row_id];
  const int nblk = (count + 31) >> 5;

  const long qrow = (long)h * NN + qbase + w * 16 + l16;
  v16h aQ[4];
#pragma unroll
  for (int ks = 0; ks < 4; ++ks) {
    v8h lo = *(const v8h*)(qb + qrow * HD + ks * 32 + o1);
    v8h hi = *(const v8h*)(qb + qrow * HD + ks * 32 + o2);
    aQ[ks] = cat16(lo, hi);
  }

  v8f oacc[8];
#pragma unroll
  for (int dt = 0; dt < 8; ++dt) oacc[dt] = (v8f){};
  float mrun[8], lrun[8];
#pragma unroll
  for (int v = 0; v < 8; ++v) { mrun[v] = -3.0e38f; lrun[v] = 0.f; }

  for (int b = 0; b < nblk; ++b) {
    if (tid < 32) {
      int i = b * 32 + tid;
      sCol[tid] = (i < count) ? idx_g[(long)row_id * NN + i] : 0;
    }
    __syncthreads();
    for (int c = tid; c < 512; c += TPB) {  // gather selected K/V columns
      int slot = c >> 4, d0 = (c & 15) * 8;
      int col = sCol[slot];
      v8h kv = *(const v8h*)(kb + ((long)h * NN + col) * HD + d0);
#pragma unroll
      for (int j = 0; j < 8; ++j) sKt[(d0 + j) * KTPAD + slot] = kv[j];
      v8h vv = *(const v8h*)(vb + ((long)h * NN + col) * HD + d0);
      *(v8h*)(sV + slot * VPAD + d0) = vv;
    }
    __syncthreads();

    v8f c0 = (v8f){}, c1 = (v8f){};
#pragma unroll
    for (int ks = 0; ks < 4; ++ks) {
      const h16* kp = sKt + ks * 32 * KTPAD + l16;
      v16h bK0, bK1;
#pragma unroll
      for (int e = 0; e < 8; ++e) {
        bK0[e] = kp[(o1 + e) * KTPAD];      bK0[8 + e] = kp[(o2 + e) * KTPAD];
        bK1[e] = kp[(o1 + e) * KTPAD + 16]; bK1[8 + e] = kp[(o2 + e) * KTPAD + 16];
      }
      c0 = wmma_f16(aQ[ks], bK0, c0);
      c1 = wmma_f16(aQ[ks], bK1, c1);
    }
    // mask padded tail slots
    bool inv0 = (b * 32 + l16) >= count;
    bool inv1 = (b * 32 + 16 + l16) >= count;
#pragma unroll
    for (int v = 0; v < 8; ++v) {
      if (inv0) c0[v] = -3.0e38f;
      if (inv1) c1[v] = -3.0e38f;
    }

    float al[8], p0[8], p1[8];
#pragma unroll
    for (int v = 0; v < 8; ++v) {
      float t = fmaxf(c0[v], c1[v]);
      t = fmaxf(t, __shfl_xor(t, 1, 32));
      t = fmaxf(t, __shfl_xor(t, 2, 32));
      t = fmaxf(t, __shfl_xor(t, 4, 32));
      t = fmaxf(t, __shfl_xor(t, 8, 32));
      float mn = fmaxf(mrun[v], t);
      al[v] = __expf(mrun[v] - mn);
      mrun[v] = mn;
      p0[v] = __expf(c0[v] - mn);
      p1[v] = __expf(c1[v] - mn);
      float rs = p0[v] + p1[v];
      rs += __shfl_xor(rs, 1, 32);
      rs += __shfl_xor(rs, 2, 32);
      rs += __shfl_xor(rs, 4, 32);
      rs += __shfl_xor(rs, 8, 32);
      lrun[v] = lrun[v] * al[v] + rs;
    }
#pragma unroll
    for (int dt = 0; dt < 8; ++dt)
#pragma unroll
      for (int v = 0; v < 8; ++v) oacc[dt][v] *= al[v];

    h16* pw = sP + w * 512;
#pragma unroll
    for (int v = 0; v < 8; ++v) {
      pw[(v + 8 * half) * 32 + l16]      = (h16)p0[v];
      pw[(v + 8 * half) * 32 + 16 + l16] = (h16)p1[v];
    }
    const h16* pp = pw + l16 * 32;
    v16h aP = cat16(*(const v8h*)(pp + o1), *(const v8h*)(pp + o2));
#pragma unroll
    for (int dt = 0; dt < 8; ++dt) {
      const h16* vp = sV + dt * 16 + l16;
      v16h bV;
#pragma unroll
      for (int e = 0; e < 8; ++e) { bV[e] = vp[(o1 + e) * VPAD]; bV[8 + e] = vp[(o2 + e) * VPAD]; }
      oacc[dt] = wmma_f16(aP, bV, oacc[dt]);
    }
    __syncthreads();
  }

  float inv[8];
#pragma unroll
  for (int v = 0; v < 8; ++v) inv[v] = 1.f / lrun[v];
#pragma unroll
  for (int dt = 0; dt < 8; ++dt)
#pragma unroll
    for (int v = 0; v < 8; ++v) {
      long r = (long)h * NN + qbase + w * 16 + v + 8 * half;
      long idx = r * HD + dt * 16 + l16;
      out_cache[idx] = o_dense[idx] - oacc[dt][v] * inv[v];
    }
}

// ---------------------------------------------------------------- launch ----
extern "C" void kernel_launch(void* const* d_in, const int* in_sizes, int n_in,
                              void* d_out, int out_size, void* d_ws, size_t ws_size,
                              hipStream_t stream) {
  const float* q      = (const float*)d_in[0];
  const float* k      = (const float*)d_in[1];
  const float* v      = (const float*)d_in[2];
  const float* prev_l = (const float*)d_in[3];
  const int*   topk   = (const int*)d_in[4];
  float* out = (float*)d_out;

  const size_t nel = (size_t)NH * NN * HD;  // 6,291,456 per tensor
  char* ws = (char*)d_ws;
  h16* qb = (h16*)ws;  ws += nel * sizeof(h16);
  h16* kb = (h16*)ws;  ws += nel * sizeof(h16);
  h16* vb = (h16*)ws;  ws += nel * sizeof(h16);
  float* cs = (float*)ws;  ws += (size_t)NH * NGRP * NN * sizeof(float);
  int* counts = (int*)ws;  ws += 256 * sizeof(int) * 4;  // padded
  int* idxg = (int*)ws;

  const int n4 = (int)(nel / 4);
  k_convert<<<(n4 + 255) / 256, 256, 0, stream>>>(q, k, v, qb, kb, vb, n4);
  k_zero_counts<<<1, 256, 0, stream>>>(counts);
  k_pass1<<<NH * NGRP, TPB, 0, stream>>>(qb, kb, vb, prev_l, out, cs);
  k_select<<<NH * NGRP, 256, 0, stream>>>(cs, topk, counts, idxg);
  k_sparse<<<NH * NGRP, TPB, 0, stream>>>(qb, kb, vb, counts, idxg, out,
                                          out + nel);
}
